// ResidualBinaryNet_49022756716720
// MI455X (gfx1250) — compile-verified, matrix operand order checked by reading, not run on hardware
//
#include <hip/hip_runtime.h>

// ---------- types ----------
typedef __bf16 bf16_t;
typedef __attribute__((ext_vector_type(16))) __bf16 v16bf;
typedef __attribute__((ext_vector_type(8)))  float  v8f;

#define V_TH   1.0f
#define BN_EPS 1e-5f
#define COUT   64
#define CONV_BM 128

__device__ __forceinline__ float  bf2f(bf16_t h) { return (float)h; }
__device__ __forceinline__ bf16_t f2bf(float f)  { return (bf16_t)f; }

// ---------- utility kernels ----------
__global__ void k_zero_f32(float* p, int n) {
    int i = blockIdx.x * blockDim.x + threadIdx.x;
    if (i < n) p[i] = 0.f;
}

// src: [Cout, Cin, ks, ks] f32 -> dst: [Cout][Kpad] bf16 (zero padded),
// k = (ky*ks + kx)*Cin + ci, Kpad = ceil32(K)
__global__ void k_cvt_weight(const float* __restrict__ src, bf16_t* __restrict__ dst,
                             int Cout, int Cin, int ks) {
    int K  = ks * ks * Cin;
    int Kp = (K + 31) & ~31;
    int n  = Cout * Kp;
    for (int i = blockIdx.x * blockDim.x + threadIdx.x; i < n; i += gridDim.x * blockDim.x) {
        int co = i / Kp, k = i % Kp;
        bf16_t v = (bf16_t)0.0f;
        if (k < K) {
            int ci = k % Cin; int r = k / Cin;
            int kx = r % ks;  int ky = r / ks;
            v = f2bf(src[((co * Cin + ci) * ks + ky) * ks + kx]);
        }
        dst[i] = v;
    }
}

// x: [B,T,C,H,W] f32 -> dst bf16 [T*B, H, W, C] (row index = t*B + b)
__global__ void k_transpose_in(const float* __restrict__ x, bf16_t* __restrict__ dst,
                               int B, int T, int C, int H, int W) {
    long long n = (long long)B * T * C * H * W;
    for (long long i = blockIdx.x * (long long)blockDim.x + threadIdx.x; i < n;
         i += (long long)gridDim.x * blockDim.x) {
        long long r = i;
        int c = (int)(r % C); r /= C;
        int w = (int)(r % W); r /= W;
        int h = (int)(r % H); r /= H;
        int b = (int)(r % B); int t = (int)(r / B);
        dst[i] = f2bf(x[((((long long)b * T + t) * C + c) * H + h) * W + w]);
    }
}

// ---------- WMMA implicit-GEMM conv (NHWC, Cout=64), compile-time shape ----------
// Double-buffered LDS pipeline: async-DMA B tile + vectorized A gather for chunk k+1
// overlap the WMMAs of chunk k; one barrier per chunk.
// in : [TB, H, W, CIN] bf16 ; wt : [64][KP] bf16 zero-padded ; out: [TB*H*W, 64] f32
template <int CIN, int KS, int PAD>
__global__ __launch_bounds__(256)
void k_conv_wmma(const bf16_t* __restrict__ in, const bf16_t* __restrict__ wt,
                 float* __restrict__ out, int TB, int H, int W) {
    constexpr int K  = KS * KS * CIN;
    constexpr int KP = (K + 31) & ~31;

    const int tid  = threadIdx.x;
    const int lane = tid & 31;
    const int wave = tid >> 5;
    const int M    = TB * H * W;
    const int m0   = blockIdx.x * CONV_BM;

    __shared__ __attribute__((aligned(32))) bf16_t Atile[2][CONV_BM][48];
    __shared__ __attribute__((aligned(32))) bf16_t Btile[2][COUT][48];
    __shared__ int rTB[CONV_BM], rIH[CONV_BM], rIW[CONV_BM];

    if (tid < CONV_BM) {
        int p = m0 + tid;
        if (p >= M) p = M - 1;
        int tb  = p / (H * W);
        int rem = p - tb * (H * W);
        rTB[tid] = tb; rIH[tid] = rem / W; rIW[tid] = rem % W;
    }
    __syncthreads();

    // ---- B tile: async DMA straight into LDS (64 rows x 64B, zero-padded wt) ----
    auto stageB = [&](int kk, int buf) {
        const int co = tid >> 2, q = tid & 3;                // 256 thr = 64 rows x 4 x 16B
        unsigned lds = (unsigned)(uintptr_t)&Btile[buf][co][q * 8];
        unsigned long long ga =
            (unsigned long long)(uintptr_t)(wt + (long long)co * KP + kk + q * 8);
        asm volatile("global_load_async_to_lds_b128 %0, %1, off"
                     :: "v"(lds), "v"(ga) : "memory");
    };

    // ---- A tile: im2col gather ----
    auto stageA = [&](int kk, int buf) {
        if constexpr ((CIN & 31) == 0) {
            // one (ky,kx) tap per 32-wide chunk: rows are contiguous 64B runs
            const int r  = kk / CIN;
            const int kx = r % KS, ky = r / KS;
            const int cb = kk % CIN;
            #pragma unroll
            for (int s = tid; s < CONV_BM * 4; s += 256) {   // (row, 16B quarter)
                const int row = s >> 2, q = s & 3;
                const int ih = rIH[row] + ky - PAD;
                const int iw = rIW[row] + kx - PAD;
                int4 val = make_int4(0, 0, 0, 0);
                if (ih >= 0 && ih < H && iw >= 0 && iw < W)
                    val = *(const int4*)&in[((((long long)rTB[row] * H + ih) * W + iw) * CIN)
                                            + cb + q * 8];
                *(int4*)&Atile[buf][row][q * 8] = val;
            }
        } else {
            // small-Cin path (first layer): scalar gather with full k decode
            const int kc = tid & 31;
            const int k  = kk + kc;
            const bool kv = (k < K);
            int ci = 0, ky = 0, kx = 0;
            if (kv) { ci = k % CIN; int r = k / CIN; kx = r % KS; ky = r / KS; }
            const int rbase = tid >> 5;
            #pragma unroll
            for (int j = 0; j < CONV_BM / 8; ++j) {
                const int row = rbase + j * 8;
                bf16_t vA = (bf16_t)0.0f;
                if (kv) {
                    int ih = rIH[row] + ky - PAD;
                    int iw = rIW[row] + kx - PAD;
                    if (ih >= 0 && ih < H && iw >= 0 && iw < W)
                        vA = in[((((long long)rTB[row] * H + ih) * W + iw) * CIN) + ci];
                }
                Atile[buf][row][kc] = vA;
            }
        }
    };

    v8f acc[4] = {};

    // prologue: fill buffer 0
    stageB(0, 0);
    stageA(0, 0);
    asm volatile("s_wait_asynccnt 0x0" ::: "memory");
    __syncthreads();

    const int mrow = wave * 16 + (lane & 15);
    const int kb   = (lane >> 4) * 16;

    int cur = 0;
    for (int kk = 0; kk < KP; kk += 32) {
        const int nxt = cur ^ 1;
        if (kk + 32 < KP) {              // prefetch next chunk into the other buffer
            stageB(kk + 32, nxt);
            stageA(kk + 32, nxt);
        }

        // fragments in contiguous k order (same for A and B, so the product is right):
        // A 16x32: lane&15 -> M row, (lane>>4)*16 -> start K (16 contiguous halves)
        // B 32x16: lane&15 -> N col, (lane>>4)*16 -> start K
        v16bf a = *(const v16bf*)&Atile[cur][mrow][kb];
        #pragma unroll
        for (int nt = 0; nt < 4; ++nt) {
            v16bf b = *(const v16bf*)&Btile[cur][nt * 16 + (lane & 15)][kb];
            acc[nt] = __builtin_amdgcn_wmma_f32_16x16x32_bf16(
                false, a, false, b, (short)0, acc[nt], false, false);
        }

        asm volatile("s_wait_asynccnt 0x0" ::: "memory");   // next B tile landed
        __syncthreads();                                    // next A tile visible, cur reads done
        cur = nxt;
    }

    // C/D 16x16 f32: lane&15 -> N, rows r + (lane>>4)*8, r=0..7 in acc VGPRs
    const int colb = lane & 15;
    const int rofs = (lane >> 4) * 8;
    #pragma unroll
    for (int nt = 0; nt < 4; ++nt) {
        #pragma unroll
        for (int r = 0; r < 8; ++r) {
            int m = m0 + wave * 16 + rofs + r;
            if (m < M) out[(long long)m * COUT + nt * 16 + colb] = acc[nt][r];
        }
    }
}

// ---------- per-channel BN statistics (sum, sumsq) ----------
__global__ __launch_bounds__(256)
void k_stats(const float* __restrict__ y, long long n, float* __restrict__ stats) {
    __shared__ float ssum[64], ssq[64];
    int tid = threadIdx.x;
    if (tid < 64) { ssum[tid] = 0.f; ssq[tid] = 0.f; }
    __syncthreads();
    float s = 0.f, q = 0.f;
    int c = tid & 63;  // channel invariant: stride is a multiple of 64
    for (long long i = (long long)blockIdx.x * blockDim.x + tid; i < n;
         i += (long long)gridDim.x * blockDim.x) {
        float v = y[i]; s += v; q += v * v;
    }
    atomicAdd(&ssum[c], s);
    atomicAdd(&ssq[c], q);
    __syncthreads();
    if (tid < 64) {
        atomicAdd(&stats[tid], ssum[tid]);
        atomicAdd(&stats[64 + tid], ssq[tid]);
    }
}

// ---------- fused BN + LIF (sequential over T, bf16 spikes out) ----------
__global__ __launch_bounds__(256)
void k_bn_lif(const float* __restrict__ y, const float* __restrict__ st,
              const float* __restrict__ gam, const float* __restrict__ bet,
              bf16_t* __restrict__ s_out, int T, long long perT, float invCnt) {
    for (long long i = (long long)blockIdx.x * blockDim.x + threadIdx.x; i < perT;
         i += (long long)gridDim.x * blockDim.x) {
        int c = (int)(i & 63);
        float mean = st[c] * invCnt;
        float var  = st[64 + c] * invCnt - mean * mean;
        float sc   = gam[c] * rsqrtf(var + BN_EPS);
        float sh   = bet[c] - mean * sc;
        float v = 0.f;
        for (int t = 0; t < T; ++t) {
            float x = y[(long long)t * perT + i] * sc + sh;
            v = 0.5f * (v + x);                    // v += (x - v)/tau, tau=2
            float spk = (v >= V_TH) ? 1.f : 0.f;
            v = (spk > 0.f) ? 0.f : v;             // hard reset
            s_out[(long long)t * perT + i] = f2bf(spk);
        }
    }
}

// ---------- fused BN(y2) + BN(ysk) + add + LIF ----------
__global__ __launch_bounds__(256)
void k_bn2_add_lif(const float* __restrict__ y2,  const float* __restrict__ st2,
                   const float* __restrict__ g2,  const float* __restrict__ b2,
                   const float* __restrict__ ysk, const float* __restrict__ stsk,
                   const float* __restrict__ gs,  const float* __restrict__ bs,
                   bf16_t* __restrict__ s_out, int T, long long perT, float invCnt) {
    for (long long i = (long long)blockIdx.x * blockDim.x + threadIdx.x; i < perT;
         i += (long long)gridDim.x * blockDim.x) {
        int c = (int)(i & 63);
        float m2 = st2[c] * invCnt,  v2 = st2[64 + c] * invCnt - m2 * m2;
        float sc2 = g2[c] * rsqrtf(v2 + BN_EPS), sh2 = b2[c] - m2 * sc2;
        float ms = stsk[c] * invCnt, vs = stsk[64 + c] * invCnt - ms * ms;
        float scs = gs[c] * rsqrtf(vs + BN_EPS), shs = bs[c] - ms * scs;
        float v = 0.f;
        for (int t = 0; t < T; ++t) {
            long long o = (long long)t * perT + i;
            float x = y2[o] * sc2 + sh2 + ysk[o] * scs + shs;
            v = 0.5f * (v + x);
            float spk = (v >= V_TH) ? 1.f : 0.f;
            v = (spk > 0.f) ? 0.f : v;
            s_out[o] = f2bf(spk);
        }
    }
}

// ---------- 2x2 maxpool on bf16 spikes (NHWC, C=64) ----------
__global__ __launch_bounds__(256)
void k_maxpool(const bf16_t* __restrict__ s, bf16_t* __restrict__ o, int TB, int H, int W) {
    int Ho = H >> 1, Wo = W >> 1;
    long long n = (long long)TB * Ho * Wo * 64;
    for (long long i = blockIdx.x * (long long)blockDim.x + threadIdx.x; i < n;
         i += (long long)gridDim.x * blockDim.x) {
        int c = (int)(i & 63); long long r = i >> 6;
        int wo = (int)(r % Wo); r /= Wo;
        int ho = (int)(r % Ho); int tb = (int)(r / Ho);
        long long base = ((((long long)tb * H + ho * 2) * W) + wo * 2) * 64 + c;
        float m = bf2f(s[base]);
        m = fmaxf(m, bf2f(s[base + 64]));
        m = fmaxf(m, bf2f(s[base + (long long)W * 64]));
        m = fmaxf(m, bf2f(s[base + (long long)W * 64 + 64]));
        o[i] = f2bf(m);
    }
}

// ---------- FC: [128 rows, D=4096] x [2, D] (reference d = c*64 + hw) ----------
__global__ __launch_bounds__(256)
void k_fc(const bf16_t* __restrict__ s, const float* __restrict__ fw,
          const float* __restrict__ fb, float* __restrict__ o) {
    __shared__ float r0[256], r1[256];
    int row = blockIdx.x, tid = threadIdx.x;
    float a0 = 0.f, a1 = 0.f;
    for (int d = tid; d < 4096; d += 256) {
        float v = bf2f(s[(long long)row * 4096 + d]);
        int c = d & 63, hw = d >> 6;
        int dr = c * 64 + hw;       // NHWC flat -> reference NCHW flat
        a0 += v * fw[dr];
        a1 += v * fw[4096 + dr];
    }
    r0[tid] = a0; r1[tid] = a1; __syncthreads();
    for (int st = 128; st > 0; st >>= 1) {
        if (tid < st) { r0[tid] += r0[tid + st]; r1[tid] += r1[tid + st]; }
        __syncthreads();
    }
    if (tid == 0) { o[row * 2 + 0] = r0[0] + fb[0]; o[row * 2 + 1] = r1[0] + fb[1]; }
}

// ---------- final LIF over T + sum ----------
__global__ void k_lif_sum(const float* __restrict__ fc, float* __restrict__ out, int T, int B) {
    int i = blockIdx.x * blockDim.x + threadIdx.x;
    if (i >= B * 2) return;
    int b = i >> 1, o = i & 1;
    float v = 0.f, acc = 0.f;
    for (int t = 0; t < T; ++t) {
        float x = fc[((t * B) + b) * 2 + o];
        v = 0.5f * (v + x);
        if (v >= V_TH) { acc += 1.f; v = 0.f; }
    }
    out[i] = acc;
}

// ---------- host orchestration ----------
extern "C" void kernel_launch(void* const* d_in, const int* in_sizes, int n_in,
                              void* d_out, int out_size, void* d_ws, size_t ws_size,
                              hipStream_t stream) {
    (void)in_sizes; (void)n_in; (void)out_size; (void)ws_size;
    const int B = 16, T = 8, TB = 128;
    const float* x   = (const float*)d_in[0];
    const float* fcw = (const float*)d_in[28];
    const float* fcb = (const float*)d_in[29];

    char* ws = (char*)d_ws;
    size_t off = 0;
    auto alloc = [&](size_t bytes) -> void* {
        void* p = ws + off;
        off += (bytes + 255) & ~(size_t)255;
        return p;
    };
    auto kpad = [](int K) { return (K + 31) & ~31; };

    const int CinB[3] = {2, 64, 64};
    bf16_t* wt_c1[3]; bf16_t* wt_c2[3]; bf16_t* wt_sk[3];
    for (int bl = 0; bl < 3; ++bl) {
        wt_c1[bl] = (bf16_t*)alloc((size_t)64 * kpad(9 * CinB[bl]) * sizeof(bf16_t));
        wt_c2[bl] = (bf16_t*)alloc((size_t)64 * kpad(9 * 64)       * sizeof(bf16_t));
        wt_sk[bl] = (bf16_t*)alloc((size_t)64 * kpad(CinB[bl])     * sizeof(bf16_t));
    }
    bf16_t* XA   = (bf16_t*)alloc((size_t)8388608  * sizeof(bf16_t)); // activations in (bf16)
    float*  Y1   = (float*) alloc((size_t)33554432 * sizeof(float));  // conv1 out
    float*  Y2   = (float*) alloc((size_t)33554432 * sizeof(float));  // conv2 out
    float*  YSK  = (float*) alloc((size_t)33554432 * sizeof(float));  // skip out
    bf16_t* S1   = (bf16_t*)alloc((size_t)33554432 * sizeof(bf16_t)); // spikes mid
    bf16_t* SOUT = (bf16_t*)alloc((size_t)33554432 * sizeof(bf16_t)); // spikes out
    float*  ST   = (float*) alloc(3 * 128 * sizeof(float));           // st1, st2, stsk
    float*  FCO  = (float*) alloc(256 * sizeof(float));

    // convert conv weights once (fp32 -> bf16, [Cout][Kpad] zero-padded, K-contiguous)
    for (int bl = 0; bl < 3; ++bl) {
        const float* c1 = (const float*)d_in[1 + bl * 9 + 0];
        const float* c2 = (const float*)d_in[1 + bl * 9 + 1];
        const float* sk = (const float*)d_in[1 + bl * 9 + 2];
        int n1 = 64 * kpad(9 * CinB[bl]), n2 = 64 * kpad(9 * 64), ns = 64 * kpad(CinB[bl]);
        k_cvt_weight<<<(n1 + 255) / 256, 256, 0, stream>>>(c1, wt_c1[bl], 64, CinB[bl], 3);
        k_cvt_weight<<<(n2 + 255) / 256, 256, 0, stream>>>(c2, wt_c2[bl], 64, 64, 3);
        k_cvt_weight<<<(ns + 255) / 256, 256, 0, stream>>>(sk, wt_sk[bl], 64, CinB[bl], 1);
    }

    // input: [B,T,2,64,64] f32 -> [T*B,64,64,2] bf16
    k_transpose_in<<<4096, 256, 0, stream>>>(x, XA, B, T, 2, 64, 64);

    const int Hs[3] = {64, 32, 16};
    for (int bl = 0; bl < 3; ++bl) {
        const int H = Hs[bl], W = Hs[bl];
        const long long P     = (long long)TB * H * W;   // per-channel BN count
        const long long total = P * 64;
        const long long perT  = total / T;
        const float invCnt = 1.f / (float)P;
        const float* g1 = (const float*)d_in[1 + bl * 9 + 3];
        const float* b1 = (const float*)d_in[1 + bl * 9 + 4];
        const float* g2 = (const float*)d_in[1 + bl * 9 + 5];
        const float* b2 = (const float*)d_in[1 + bl * 9 + 6];
        const float* gs = (const float*)d_in[1 + bl * 9 + 7];
        const float* bs = (const float*)d_in[1 + bl * 9 + 8];
        float* st1 = ST; float* st2 = ST + 128; float* stsk = ST + 256;

        const int gconv = (int)(P / CONV_BM);

        k_zero_f32<<<2, 256, 0, stream>>>(ST, 384);
        if (bl == 0) {
            k_conv_wmma<2, 3, 1><<<gconv, 256, 0, stream>>>(XA, wt_c1[bl], Y1, TB, H, W);
        } else {
            k_conv_wmma<64, 3, 1><<<gconv, 256, 0, stream>>>(XA, wt_c1[bl], Y1, TB, H, W);
        }
        k_stats<<<1024, 256, 0, stream>>>(Y1, total, st1);
        k_bn_lif<<<2048, 256, 0, stream>>>(Y1, st1, g1, b1, S1, T, perT, invCnt);
        k_conv_wmma<64, 3, 1><<<gconv, 256, 0, stream>>>(S1, wt_c2[bl], Y2, TB, H, W);
        k_stats<<<1024, 256, 0, stream>>>(Y2, total, st2);
        if (bl == 0) {
            k_conv_wmma<2, 1, 0><<<gconv, 256, 0, stream>>>(XA, wt_sk[bl], YSK, TB, H, W);
        } else {
            k_conv_wmma<64, 1, 0><<<gconv, 256, 0, stream>>>(XA, wt_sk[bl], YSK, TB, H, W);
        }
        k_stats<<<1024, 256, 0, stream>>>(YSK, total, stsk);
        k_bn2_add_lif<<<2048, 256, 0, stream>>>(Y2, st2, g2, b2, YSK, stsk, gs, bs,
                                                SOUT, T, perT, invCnt);
        long long npool = (long long)TB * (H / 2) * (W / 2) * 64;
        k_maxpool<<<(int)((npool + 255) / 256), 256, 0, stream>>>(SOUT, XA, TB, H, W);
    }

    k_fc<<<128, 256, 0, stream>>>(XA, fcw, fcb, FCO);
    k_lif_sum<<<1, 32, 0, stream>>>(FCO, (float*)d_out, T, B);
}
